// AttentionActPrune_17179869702
// MI455X (gfx1250) — matrix operand order repaired and checked by reading, not compiled, not run on hardware
//
#include <hip/hip_runtime.h>
#include <hip/hip_bf16.h>

// ---------------------------------------------------------------------------
// MHA forward for MI455X (gfx1250, wave32, WMMA f32_16x16x32_f16)
// B=16, S=1024, d=1024, H=16, Dh=64
// Pipeline: [cvt fp32->f16] -> [QKV gemm] -> [flash attention] -> [out gemm]
// GEMM/attention tiles staged into LDS with GLOBAL_LOAD_ASYNC_TO_LDS_B128,
// double-buffered, waited with s_wait_asynccnt.
// ---------------------------------------------------------------------------

typedef _Float16 v16h __attribute__((ext_vector_type(16)));
typedef _Float16 v8h  __attribute__((ext_vector_type(8)));
typedef float    v8f  __attribute__((ext_vector_type(8)));

static constexpr int Bb = 16;    // batch
static constexpr int Sq = 1024;  // sequence
static constexpr int D  = 1024;  // model dim
static constexpr int H  = 16;    // heads
static constexpr int DH = 64;    // head dim

static __device__ __forceinline__ v8f wmma32(v16h a, v16h b, v8f c) {
  return __builtin_amdgcn_wmma_f32_16x16x32_f16(
      false, a, false, b, (short)0, c, false, false);
}

static __device__ __forceinline__ v16h join16(v8h lo, v8h hi) {
  union { v16h v; v8h h[2]; } u;
  u.h[0] = lo; u.h[1] = hi;
  return u.v;
}

static __device__ __forceinline__ v8h cvt8(v8f x) {
  v8h r;
#pragma unroll
  for (int i = 0; i < 8; ++i) r[i] = (_Float16)x[i];
  return r;
}

// A-fragment (16x32 f16): lanes 0-15 hold K {0..15} of row l15,
// lanes 16-31 hold K {16..31}.
static __device__ __forceinline__ v16h load_a_f16(const _Float16* rowp, int hi16) {
  const int kb = hi16 ? 16 : 0;
  v8h lo = *(const v8h*)(rowp + kb);
  v8h hi = *(const v8h*)(rowp + kb + 8);
  return join16(lo, hi);
}

// ---- CDNA5 async copy to LDS (16B per lane), ASYNCcnt tracked -------------
static __device__ __forceinline__ void async_b128(const _Float16* gsrc,
                                                  _Float16* ldst) {
#if __has_builtin(__builtin_amdgcn_global_load_async_to_lds_b128)
  // Builtin signature (from hipcc diagnostic): global v4i* src, LDS v4i* dst.
  typedef int v4i __attribute__((vector_size(16)));
  typedef __attribute__((address_space(1))) v4i* gvp;
  typedef __attribute__((address_space(3))) v4i* lvp;
  __builtin_amdgcn_global_load_async_to_lds_b128((gvp)gsrc, (lvp)ldst, 0, 0);
#else
  unsigned loff =
      (unsigned)(size_t)(__attribute__((address_space(3))) _Float16*)ldst;
  asm volatile("global_load_async_to_lds_b128 %0, %1, off"
               :: "v"(loff), "v"(gsrc) : "memory");
#endif
}

template <int N>
static __device__ __forceinline__ void wait_async() {
  asm volatile("s_wait_asynccnt %0" :: "n"(N) : "memory");
}

// ---------------------------------------------------------------------------
// Kernel 0: fp32 -> f16 conversion (bandwidth-bound, one pass).
// ---------------------------------------------------------------------------
__global__ __launch_bounds__(256) void cvt_f16_kernel(
    const float* __restrict__ src, _Float16* __restrict__ dst) {
  const size_t i = ((size_t)blockIdx.x * 256 + threadIdx.x) * 8;
  v8f x = *(const v8f*)(src + i);
  *(v8h*)(dst + i) = cvt8(x);
}

// ---------------------------------------------------------------------------
// Kernel 1: fused QKV projection.  y = x @ W^T + b  (blockIdx.z selects W)
// 256 thr = 8 waves; tile M=64 x N=128; wave tile 16x64.
// B tile (128n x 32k, f16, 8KB) staged in LDS [n][k] via async DMA,
// double-buffered; A-fragments read directly from pre-converted Xh.
// ---------------------------------------------------------------------------
__global__ __launch_bounds__(256) void qkv_proj_kernel(
    const _Float16* __restrict__ Xh,
    const _Float16* __restrict__ Wqh, const float* __restrict__ bq,
    const _Float16* __restrict__ Wkh, const float* __restrict__ bk,
    const _Float16* __restrict__ Wvh, const float* __restrict__ bv,
    _Float16* __restrict__ Qw, _Float16* __restrict__ Kw,
    _Float16* __restrict__ Vtw) {
  const int tid  = threadIdx.x;
  const int wave = tid >> 5;
  const int lane = tid & 31;
  const int l15  = lane & 15;
  const int hi16 = lane & 16;
  const int wm   = wave >> 1;
  const int wn   = wave & 1;

  const int m0 = blockIdx.x * 64 + wm * 16;
  const int n0 = blockIdx.y * 128 + wn * 64;
  const int z  = blockIdx.z;

  const _Float16* Wh  = (z == 0) ? Wqh : (z == 1) ? Wkh : Wvh;
  const float* bias   = (z == 0) ? bq : (z == 1) ? bk : bv;

  __shared__ __align__(64) _Float16 bsh[2][128 * 32];  // 2 x 8KB

  const _Float16* ax = Xh + (size_t)(m0 + l15) * D;
  const int kd = hi16 ? 16 : 0;

  auto stage = [&](int i) {
    const int k0 = i * 32;
    _Float16* dst = &bsh[i & 1][0];
#pragma unroll
    for (int c = 0; c < 2; ++c) {
      const int o  = (c * 256 + tid) * 8;  // half index, 16B chunks
      const int n  = o >> 5;               // 32 halfs per LDS row
      const int kk = o & 31;
      async_b128(Wh + (size_t)(n0 + n) * D + k0 + kk, dst + o);
    }
  };

  v8f acc[4] = {};
  stage(0);
  for (int i = 0; i < 32; ++i) {
    if (i < 31) { stage(i + 1); wait_async<2>(); } else { wait_async<0>(); }
    __syncthreads();
    const _Float16* bbuf = &bsh[i & 1][0];
    const v16h a = load_a_f16(ax + i * 32, hi16);
#pragma unroll
    for (int t = 0; t < 4; ++t) {
      const v16h bf = *(const v16h*)(bbuf + (t * 16 + l15) * 32 + kd);
      acc[t] = wmma32(a, bf, acc[t]);
    }
    __syncthreads();
  }

  // Epilogue: bias + scatter into per-head layouts.
  const int mbase = m0 + (hi16 ? 8 : 0);
#pragma unroll
  for (int t = 0; t < 4; ++t) {
    const int n = n0 + t * 16 + l15;
    const float bv_ = bias[n];
    const int h  = n >> 6;
    const int dd = n & 63;
#pragma unroll
    for (int r = 0; r < 8; ++r) {
      const int m = mbase + r;
      const int b = m >> 10;
      const int s = m & 1023;
      const float val = acc[t][r] + bv_;
      const size_t head = (size_t)b * H + h;
      if (z == 0) {
        Qw[(head * Sq + s) * DH + dd] = (_Float16)(val * 0.125f);  // 1/sqrt(Dh)
      } else if (z == 1) {
        Kw[(head * Sq + s) * DH + dd] = (_Float16)val;
      } else {
        Vtw[(head * DH + dd) * Sq + s] = (_Float16)val;            // transposed
      }
    }
  }
}

// ---------------------------------------------------------------------------
// Kernel 2: flash attention per (b,h).  Grid (16,16,16), 128 thr = 4 waves.
// K tile [32 keys][64 dims] and V^T tile [64 dims][32 keys] staged in LDS via
// async DMA (shared by all 4 waves), double-buffered.  Each wave owns 16
// query rows; P goes through a wave-private LDS transpose to A-layout.
// ---------------------------------------------------------------------------
__global__ __launch_bounds__(128) void attention_kernel(
    const _Float16* __restrict__ Qw, const _Float16* __restrict__ Kw,
    const _Float16* __restrict__ Vtw, _Float16* __restrict__ Ctx) {
  const int tid  = threadIdx.x;
  const int wave = tid >> 5;
  const int lane = tid & 31;
  const int l15  = lane & 15;
  const int hi16 = lane & 16;

  const int b = blockIdx.z;
  const int h = blockIdx.y;
  const int q0 = blockIdx.x * 64 + wave * 16;

  const size_t head = (size_t)b * H + h;
  const _Float16* Qp = Qw + head * Sq * DH;
  const _Float16* Kp = Kw + head * Sq * DH;
  const _Float16* Vp = Vtw + head * DH * Sq;

  __shared__ __align__(64) _Float16 kbuf[2][32 * 64];  // 2 x 4KB  [key][dim]
  __shared__ __align__(64) _Float16 vbuf[2][64 * 32];  // 2 x 4KB  [dim][key]
  __shared__ __align__(64) _Float16 psh[4][16][32];    // per-wave P staging

  // Loop-invariant Q A-fragments (Dh=64 -> two 16x32 frags), pre-scaled.
  const _Float16* qrow = Qp + (size_t)(q0 + l15) * DH;
  const v16h a0 = load_a_f16(qrow, hi16);
  const v16h a1 = load_a_f16(qrow + 32, hi16);

  float mrun[8], lrun[8];
  v8f acc[4] = {};
#pragma unroll
  for (int r = 0; r < 8; ++r) { mrun[r] = -1e30f; lrun[r] = 0.f; }

  const int kd = hi16 ? 16 : 0;

  auto stage = [&](int i) {
    const int kt = i * 32;
#pragma unroll
    for (int c = 0; c < 2; ++c) {            // K tile: 2 x 2KB
      const int o   = (c * 128 + tid) * 8;
      const int key = o >> 6;                // 64 halfs per key row
      const int db  = o & 63;
      async_b128(Kp + (size_t)(kt + key) * DH + db, &kbuf[i & 1][o]);
    }
#pragma unroll
    for (int c = 0; c < 2; ++c) {            // V tile: 2 x 2KB
      const int o  = (c * 128 + tid) * 8;
      const int d  = o >> 5;                 // 32 halfs per dim row
      const int kb = o & 31;
      async_b128(Vp + (size_t)d * Sq + kt + kb, &vbuf[i & 1][o]);
    }
  };

  stage(0);
  for (int i = 0; i < 32; ++i) {
    if (i < 31) { stage(i + 1); wait_async<4>(); } else { wait_async<0>(); }
    __syncthreads();
    const _Float16* kb_ = &kbuf[i & 1][0];
    const _Float16* vb_ = &vbuf[i & 1][0];

    // ---- scores for two 16-key tiles (from LDS) ----
    v8f s0 = {}, s1 = {};
    s0 = wmma32(a0, *(const v16h*)(kb_ + l15 * 64 + kd), s0);
    s0 = wmma32(a1, *(const v16h*)(kb_ + l15 * 64 + 32 + kd), s0);
    s1 = wmma32(a0, *(const v16h*)(kb_ + (16 + l15) * 64 + kd), s1);
    s1 = wmma32(a1, *(const v16h*)(kb_ + (16 + l15) * 64 + 32 + kd), s1);

    // ---- online softmax (rows live across 16-lane half-waves) ----
#pragma unroll
    for (int r = 0; r < 8; ++r) {
      float t = fmaxf(s0[r], s1[r]);
#pragma unroll
      for (int off = 8; off >= 1; off >>= 1)
        t = fmaxf(t, __shfl_xor(t, off, 16));
      const float mnew = fmaxf(mrun[r], t);
      const float sc = __expf(mrun[r] - mnew);
      mrun[r] = mnew;
      const float p0 = __expf(s0[r] - mnew);
      const float p1 = __expf(s1[r] - mnew);
      lrun[r] = lrun[r] * sc + p0 + p1;
      acc[0][r] *= sc; acc[1][r] *= sc; acc[2][r] *= sc; acc[3][r] *= sc;
      const int row = r + (hi16 ? 8 : 0);
      psh[wave][row][l15]      = (_Float16)p0;
      psh[wave][row][l15 + 16] = (_Float16)p1;
    }

    // ---- wave-private LDS transpose: P D-layout -> A-layout ----
    const _Float16* pr = &psh[wave][l15][0];
    const int kb = hi16 ? 16 : 0;
    const v16h pa = join16(*(const v8h*)(pr + kb), *(const v8h*)(pr + kb + 8));

    // ---- ctx += P @ V  (V^T rows contiguous in keys, from LDS) ----
#pragma unroll
    for (int t4 = 0; t4 < 4; ++t4) {
      const v16h vf = *(const v16h*)(vb_ + (t4 * 16 + l15) * 32 + kd);
      acc[t4] = wmma32(pa, vf, acc[t4]);
    }
    __syncthreads();
  }

  // ---- finalize: full row sums, normalize, store ctx f16 [B,S,H*Dh] ----
  float inv[8];
#pragma unroll
  for (int r = 0; r < 8; ++r) {
    float L = lrun[r];
#pragma unroll
    for (int off = 8; off >= 1; off >>= 1) L += __shfl_xor(L, off, 16);
    inv[r] = 1.0f / L;
  }
#pragma unroll
  for (int t4 = 0; t4 < 4; ++t4) {
    const int feat = h * DH + t4 * 16 + l15;
#pragma unroll
    for (int r = 0; r < 8; ++r) {
      const int s = q0 + r + (hi16 ? 8 : 0);
      Ctx[((size_t)b * Sq + s) * D + feat] = (_Float16)(acc[t4][r] * inv[r]);
    }
  }
}

// ---------------------------------------------------------------------------
// Kernel 3: output projection.  out = ctx @ Wo^T + bo (fp32 out), same
// async-LDS staged GEMM as kernel 1.
// ---------------------------------------------------------------------------
__global__ __launch_bounds__(256) void out_proj_kernel(
    const _Float16* __restrict__ Ctx, const _Float16* __restrict__ Woh,
    const float* __restrict__ bo, float* __restrict__ Out) {
  const int tid  = threadIdx.x;
  const int wave = tid >> 5;
  const int lane = tid & 31;
  const int l15  = lane & 15;
  const int hi16 = lane & 16;
  const int wm   = wave >> 1;
  const int wn   = wave & 1;

  const int m0 = blockIdx.x * 64 + wm * 16;
  const int n0 = blockIdx.y * 128 + wn * 64;

  __shared__ __align__(64) _Float16 bsh[2][128 * 32];

  const _Float16* ax = Ctx + (size_t)(m0 + l15) * D;
  const int kd = hi16 ? 16 : 0;

  auto stage = [&](int i) {
    const int k0 = i * 32;
    _Float16* dst = &bsh[i & 1][0];
#pragma unroll
    for (int c = 0; c < 2; ++c) {
      const int o  = (c * 256 + tid) * 8;
      const int n  = o >> 5;
      const int kk = o & 31;
      async_b128(Woh + (size_t)(n0 + n) * D + k0 + kk, dst + o);
    }
  };

  v8f acc[4] = {};
  stage(0);
  for (int i = 0; i < 32; ++i) {
    if (i < 31) { stage(i + 1); wait_async<2>(); } else { wait_async<0>(); }
    __syncthreads();
    const _Float16* bbuf = &bsh[i & 1][0];
    const v16h a = load_a_f16(ax + i * 32, hi16);
#pragma unroll
    for (int t = 0; t < 4; ++t) {
      const v16h bf = *(const v16h*)(bbuf + (t * 16 + l15) * 32 + kd);
      acc[t] = wmma32(a, bf, acc[t]);
    }
    __syncthreads();
  }

  const int mbase = m0 + (hi16 ? 8 : 0);
#pragma unroll
  for (int t = 0; t < 4; ++t) {
    const int n = n0 + t * 16 + l15;
    const float bb = bo[n];
#pragma unroll
    for (int r = 0; r < 8; ++r) {
      const int m = mbase + r;
      Out[(size_t)m * D + n] = acc[t][r] + bb;
    }
  }
}

// ---------------------------------------------------------------------------
extern "C" void kernel_launch(void* const* d_in, const int* in_sizes, int n_in,
                              void* d_out, int out_size, void* d_ws,
                              size_t ws_size, hipStream_t stream) {
  (void)in_sizes; (void)n_in; (void)out_size; (void)ws_size;

  const float* X  = (const float*)d_in[0];
  const float* Wq = (const float*)d_in[1];
  const float* bq = (const float*)d_in[2];
  const float* Wk = (const float*)d_in[3];
  const float* bk = (const float*)d_in[4];
  const float* Wv = (const float*)d_in[5];
  const float* bv = (const float*)d_in[6];
  const float* Wo = (const float*)d_in[7];
  const float* bo = (const float*)d_in[8];
  float* Out = (float*)d_out;

  // Workspace carve-up (f16 halfs):
  //   Xh (32MiB) | Wq/Wk/Wv/Wo h (4 x 2MiB) | Q | K | V^T | Ctx (4 x 32MiB)
  const size_t nX = (size_t)Bb * Sq * D;  // 16,777,216
  const size_t nW = (size_t)D * D;        //  1,048,576
  _Float16* Xh  = (_Float16*)d_ws;
  _Float16* Wqh = Xh + nX;
  _Float16* Wkh = Wqh + nW;
  _Float16* Wvh = Wkh + nW;
  _Float16* Woh = Wvh + nW;
  _Float16* Qw  = Woh + nW;
  _Float16* Kw  = Qw + nX;
  _Float16* Vtw = Kw + nX;
  _Float16* Ctx = Vtw + nX;

  // Stage 0: one-pass fp32 -> f16 of activations and all weights.
  cvt_f16_kernel<<<dim3(nX / 2048), 256, 0, stream>>>(X, Xh);
  cvt_f16_kernel<<<dim3(nW / 2048), 256, 0, stream>>>(Wq, Wqh);
  cvt_f16_kernel<<<dim3(nW / 2048), 256, 0, stream>>>(Wk, Wkh);
  cvt_f16_kernel<<<dim3(nW / 2048), 256, 0, stream>>>(Wv, Wvh);
  cvt_f16_kernel<<<dim3(nW / 2048), 256, 0, stream>>>(Wo, Woh);

  qkv_proj_kernel<<<dim3(256, 8, 3), 256, 0, stream>>>(
      Xh, Wqh, bq, Wkh, bk, Wvh, bv, Qw, Kw, Vtw);

  attention_kernel<<<dim3(Sq / 64, H, Bb), 128, 0, stream>>>(Qw, Kw, Vtw, Ctx);

  out_proj_kernel<<<dim3(256, 8, 1), 256, 0, stream>>>(Ctx, Woh, bo, Out);
}
